// QLSTMModel_20426864460372
// MI455X (gfx1250) — matrix operand order repaired and checked by reading, not compiled, not run on hardware
//
#include <hip/hip_runtime.h>

typedef __attribute__((ext_vector_type(16))) _Float16 v16h;
typedef __attribute__((ext_vector_type(8)))  _Float16 v8h;
typedef __attribute__((ext_vector_type(8)))  float    v8f;

#define BATCH 4096
#define SEQ   200
#define HID   64
#define TILE  16         // batch rows per tile (WMMA M)
#define WPB   4          // 4 waves cooperate on one batch tile; wave w owns cg=w

__device__ __forceinline__ float fsig(float x) {
    // sigmoid(x) = 1 / (1 + e^-x) -> v_exp + v_rcp
    return __builtin_amdgcn_rcpf(1.0f + __expf(-x));
}
__device__ __forceinline__ float ftanhf(float x) {
    // tanh(x) = 1 - 2/(e^{2x}+1); saturates correctly at +-1 via inf/rcp
    return 1.0f - 2.0f * __builtin_amdgcn_rcpf(__expf(2.0f * x) + 1.0f);
}

__global__ __launch_bounds__(32 * WPB) void lstm_wmma_kernel(
    const float* __restrict__ xin,   // [BATCH*SEQ]
    const float* __restrict__ W_ih,  // [256]
    const float* __restrict__ W_hh,  // [256*64] row-major (4H x H)
    const float* __restrict__ b_ih,  // [256]
    const float* __restrict__ b_hh,  // [256]
    const float* __restrict__ W_lin, // [SEQ*HID]
    const float* __restrict__ b_lin, // [1]
    float* __restrict__ out)         // [BATCH]
{
    // per-tile h transpose bounce: 16 rows x 64 cols f16 = 2 KB
    __shared__ __align__(16) _Float16 hbuf[TILE * HID];
    __shared__ float xbuf[TILE];        // x_t broadcast for the 16 batch rows
    __shared__ float obuf[WPB][TILE];   // per-wave head partials

    const int tid    = threadIdx.x;
    const int wave   = tid >> 5;        // == cg (h-column group this wave owns)
    const int lane   = tid & 31;
    const int lane16 = lane & 15;
    const int hi     = lane >> 4;
    const int cg     = wave;

    const int rowBase = blockIdx.x * TILE;

    // ---- this wave's 8 B fragments (g in 0..3, kt in 0..1) live in registers
    // for the entire sequence.  B[k][n] = W_hh[n*HID + k], f16.
    // B 32x16 f16 layout: lane16 = column N; lanes 0-15: e0..7->K0..7,
    // e8..15->K16..23; lanes 16-31: +8.
    v16h bfr[4][2];
    #pragma unroll
    for (int g = 0; g < 4; ++g) {
        const int n = (g * 4 + cg) * 16 + lane16;
        #pragma unroll
        for (int kt = 0; kt < 2; ++kt) {
            #pragma unroll
            for (int e = 0; e < 16; ++e)
                bfr[g][kt][e] = (_Float16)W_hh[n * HID + kt * 32 + hi * 8 + e + (e >= 8 ? 8 : 0)];
        }
    }

    // per-lane gate constants for this wave's columns: col = g*HID + cg*16 + lane16
    float wihC[4], bC[4];
    #pragma unroll
    for (int g = 0; g < 4; ++g) {
        const int col = g * HID + cg * 16 + lane16;
        wihC[g] = W_ih[col];
        bC[g]   = b_ih[col] + b_hh[col];
    }

    if (wave == 0 && lane < TILE) xbuf[lane] = xin[(rowBase + lane) * SEQ];

    __syncthreads();

    // cell state for this wave's columns, C-layout: cSt[v] = c[row v+hi*8][cg*16+lane16]
    v8f cSt;
    #pragma unroll
    for (int v = 0; v < 8; ++v) cSt[v] = 0.0f;

    // A fragments of h_{t-1} (h0 = 0)
    v16h a0, a1;
    #pragma unroll
    for (int e = 0; e < 16; ++e) { a0[e] = (_Float16)0.0f; a1[e] = (_Float16)0.0f; }

    float outAcc[8];
    #pragma unroll
    for (int v = 0; v < 8; ++v) outAcc[v] = 0.0f;

    // x_0 broadcast for the rows this lane's accumulator VGPRs cover
    float xr[8];
    #pragma unroll
    for (int v = 0; v < 8; ++v) xr[v] = xbuf[v + hi * 8];

    for (int t = 0; t < SEQ; ++t) {
        // gates for this wave's N-tiles {cg, cg+4, cg+8, cg+12}
        v8f acc[4];
        #pragma unroll
        for (int g = 0; g < 4; ++g) {
            #pragma unroll
            for (int v = 0; v < 8; ++v)
                acc[g][v] = fmaf(xr[v], wihC[g], bC[g]);   // x-projection + bias
            acc[g] = __builtin_amdgcn_wmma_f32_16x16x32_f16(
                false, a0, false, bfr[g][0], (short)0, acc[g], false, false);
            acc[g] = __builtin_amdgcn_wmma_f32_16x16x32_f16(
                false, a1, false, bfr[g][1], (short)0, acc[g], false, false);
        }

        // nonlinearity + state update + fused output head for our 16 columns
        const float wl = W_lin[t * HID + cg * 16 + lane16];
        #pragma unroll
        for (int v = 0; v < 8; ++v) {
            const float iv = fsig(acc[0][v]);
            const float fv = fsig(acc[1][v]);
            const float gv = ftanhf(acc[2][v]);
            const float ov = fsig(acc[3][v]);
            const float cv = fmaf(fv, cSt[v], iv * gv);
            cSt[v]         = cv;
            const float hv = ov * ftanhf(cv);
            outAcc[v] = fmaf(hv, wl, outAcc[v]);
            hbuf[(v + hi * 8) * HID + cg * 16 + lane16] = (_Float16)hv;
        }

        if (wave == 0 && lane < TILE && t + 1 < SEQ)
            xbuf[lane] = xin[(rowBase + lane) * SEQ + (t + 1)];

        __syncthreads();  // h_t columns + x_{t+1} complete across all 4 waves

        // rebuild full A fragments for h_t.
        // 16-bit A 16x32 layout: lane16 = row M; lanes 0-15: e0..7->K0..7,
        // e8..15->K16..23; lanes 16-31: +8.  K-tile kt adds kt*32.
        {
            const int rb = lane16 * HID + hi * 8;
            const v8h l0 = *(const v8h*)&hbuf[rb + 0];
            const v8h h0 = *(const v8h*)&hbuf[rb + 16];
            const v8h l1 = *(const v8h*)&hbuf[rb + 32];
            const v8h h1 = *(const v8h*)&hbuf[rb + 48];
            #pragma unroll
            for (int e = 0; e < 8; ++e) {
                a0[e] = l0[e]; a0[e + 8] = h0[e];
                a1[e] = l1[e]; a1[e + 8] = h1[e];
            }
        }
        #pragma unroll
        for (int v = 0; v < 8; ++v) xr[v] = xbuf[v + hi * 8];

        __syncthreads();  // all waves done reading before next step overwrites
    }

    // head reduction: sum over the 16 lanes of each half-wave -> per-wave row partial
    #pragma unroll
    for (int v = 0; v < 8; ++v) {
        float s = outAcc[v];
        #pragma unroll
        for (int m = 8; m >= 1; m >>= 1)
            s += __shfl_xor(s, m, 32);
        if (lane16 == 0) obuf[wave][hi * 8 + v] = s;
    }
    __syncthreads();
    if (wave == 0 && lane < TILE)
        out[rowBase + lane] = obuf[0][lane] + obuf[1][lane] +
                              obuf[2][lane] + obuf[3][lane] + b_lin[0];
}

extern "C" void kernel_launch(void* const* d_in, const int* in_sizes, int n_in,
                              void* d_out, int out_size, void* d_ws, size_t ws_size,
                              hipStream_t stream) {
    (void)in_sizes; (void)n_in; (void)d_ws; (void)ws_size; (void)out_size;
    const float* xin  = (const float*)d_in[0];
    const float* wih  = (const float*)d_in[1];
    const float* whh  = (const float*)d_in[2];
    const float* bih  = (const float*)d_in[3];
    const float* bhh  = (const float*)d_in[4];
    const float* wlin = (const float*)d_in[5];
    const float* blin = (const float*)d_in[6];
    float* out = (float*)d_out;

    dim3 grid(BATCH / TILE);     // 256 blocks (one batch tile each)
    dim3 block(32 * WPB);        // 128 threads = 4 cooperating waves
    lstm_wmma_kernel<<<grid, block, 0, stream>>>(xin, wih, whh, bih, bhh, wlin, blin, out);
}
